// MainEffectsMLP_23527830848187
// MI455X (gfx1250) — matrix operand order repaired and checked by reading, not compiled, not run on hardware
//
#include <hip/hip_runtime.h>
#include <hip/hip_bf16.h>

// MI455X / gfx1250, wave32. Compute-bound workload -> bf16 WMMA everywhere.
// Main GEMM: async global->LDS B staging (double buffered) + 32x64 wave tiles,
// branchless software pipeline (prefetch index clamped instead of guarded).

typedef __bf16 bf16;
typedef __attribute__((ext_vector_type(16))) __bf16 v16bf;
typedef __attribute__((ext_vector_type(8)))  __bf16 v8bf;
typedef __attribute__((ext_vector_type(8)))  float  v8f;
typedef __attribute__((ext_vector_type(4)))  float  f32x4;

#define BATCH 16384
#define NINP  256
#define HID   1024
#define NOUT  16
#define UHU   32

static __device__ __forceinline__ bf16 f2bf(float f) { return (bf16)f; }

static __device__ __forceinline__ v8f wmma_bf16(const v16bf& a, const v16bf& b, const v8f& c) {
  return __builtin_amdgcn_wmma_f32_16x16x32_bf16(false, a, false, b, (short)0, c, false, false);
}

// Async copy of 16B from global to LDS (one instruction per wave, per-lane addresses).
static __device__ __forceinline__ void async_ld_b128(void* lds_dst, const void* gsrc) {
  unsigned int lds_addr = (unsigned int)(unsigned long long)lds_dst; // LDS aperture: low 32 bits = LDS offset
  unsigned long long gaddr = (unsigned long long)gsrc;
  asm volatile("global_load_async_to_lds_b128 %0, %1, off"
               :: "v"(lds_addr), "v"(gaddr) : "memory");
}
static __device__ __forceinline__ void wait_async0() {
  asm volatile("s_wait_asynccnt 0x0" ::: "memory");
}

// ---------------- prep kernels ----------------

__global__ void cvt_bf16_kernel(const float* __restrict__ src, bf16* __restrict__ dst, int n) {
  int i = blockIdx.x * blockDim.x + threadIdx.x;
  if (i < n) dst[i] = f2bf(src[i]);
}

// Pack (possibly batched) row-major W[K][N] f32 into WMMA B-fragment order:
// P[b][kt][nt][lane][e] = W[b][kt*32 + (e>>3)*16 + (lane>>4)*8 + (e&7)][nt*16 + (lane&15)]
__global__ void pack_b_frag_kernel(const float* __restrict__ W, bf16* __restrict__ P,
                                   int K, int N, int total) {
  int idx = blockIdx.x * blockDim.x + threadIdx.x;
  if (idx >= total) return;
  int per  = K * N;
  int b    = idx / per;
  int r    = idx - b * per;
  int e    = r & 15;
  int lane = (r >> 4) & 31;
  int t    = r >> 9;                 // 512 elements per 32x16 tile
  int NT   = N >> 4;
  int nt   = t % NT;
  int kt   = t / NT;
  int k    = kt * 32 + ((e >> 3) << 4) + ((lane >> 4) << 3) + (e & 7);
  int n    = nt * 16 + (lane & 15);
  P[idx] = f2bf(W[(size_t)b * per + (size_t)k * N + n]);
}

__global__ void ub3_sum_kernel(const float* __restrict__ ub3, float* __restrict__ out) {
  int n = threadIdx.x;               // 16 threads
  float s = 0.f;
  for (int i = 0; i < NINP; ++i) s += ub3[i * NOUT + n];
  out[n] = s;
}

// ---------------- main GEMM: Out = act(A @ B + bias), bf16 in/out ----------------
// grid (M/256, N/64), 256 threads = 8 waves; each wave: 32(M) x 64(N) strip.
// B chunk (4KB per K-step, shared by all waves) staged in LDS via async loads,
// double buffered; A fragments double buffered in registers (branchless prefetch).

__global__ __launch_bounds__(256) void gemm_bf16_wmma(
    const bf16* __restrict__ A, const bf16* __restrict__ Bp,
    const float* __restrict__ bias, bf16* __restrict__ Out,
    int M, int N, int K, int do_relu)
{
  __shared__ bf16 ldsB[2][2048];     // 2 x 4KB: 4 n-tiles x (32 lanes x 16 bf16)

  const int tid  = threadIdx.x;
  const int lane = tid & 31;
  const int wave = tid >> 5;
  const int half = lane >> 4;
  const int mr   = lane & 15;
  const int m0   = blockIdx.x * 256 + wave * 32;
  const int n0   = blockIdx.y * 64;
  const int NT   = N >> 4;
  const int nt0  = n0 >> 4;
  const int KT   = K >> 5;

  v8f acc[2][4] = {{{}, {}, {}, {}}, {{}, {}, {}, {}}};

  const bf16* arow0 = A + (size_t)(m0 + mr) * K + half * 8;
  const bf16* arow1 = A + (size_t)(m0 + 16 + mr) * K + half * 8;

  // prologue: stage B chunk kt=0, A fragments kt=0
  async_ld_b128(&ldsB[0][tid * 8], Bp + ((size_t)0 * NT + nt0) * 512 + tid * 8);
  v8bf a_lo0 = *(const v8bf*)(arow0);
  v8bf a_hi0 = *(const v8bf*)(arow0 + 16);
  v8bf a_lo1 = *(const v8bf*)(arow1);
  v8bf a_hi1 = *(const v8bf*)(arow1 + 16);

  for (int kt = 0; kt < KT; ++kt) {
    wait_async0();                   // this wave's stage of buf[kt&1] has landed
    __syncthreads();                 // everyone's stage landed; prior readers of buf[(kt+1)&1] done

    // branchless prefetch: clamp index on the last step (redundant loads are harmless;
    // the LDS buffer being re-staged had all its readers pass the previous barrier).
    const int ktn = (kt + 1 < KT) ? (kt + 1) : kt;
    async_ld_b128(&ldsB[(kt + 1) & 1][tid * 8],
                  Bp + ((size_t)ktn * NT + nt0) * 512 + tid * 8);
    const bf16* ap0 = arow0 + ktn * 32;
    const bf16* ap1 = arow1 + ktn * 32;
    v8bf na_lo0 = *(const v8bf*)(ap0);
    v8bf na_hi0 = *(const v8bf*)(ap0 + 16);
    v8bf na_lo1 = *(const v8bf*)(ap1);
    v8bf na_hi1 = *(const v8bf*)(ap1 + 16);

    v16bf afrag0, afrag1;
#pragma unroll
    for (int e = 0; e < 8; ++e) {
      afrag0[e] = a_lo0[e]; afrag0[8 + e] = a_hi0[e];
      afrag1[e] = a_lo1[e]; afrag1[8 + e] = a_hi1[e];
    }

    const bf16* bufc = &ldsB[kt & 1][0];
#pragma unroll
    for (int t = 0; t < 4; ++t) {
      v16bf bfrag = *(const v16bf*)(bufc + (t * 32 + lane) * 16);
      acc[0][t] = wmma_bf16(afrag0, bfrag, acc[0][t]);
      acc[1][t] = wmma_bf16(afrag1, bfrag, acc[1][t]);
    }

    a_lo0 = na_lo0; a_hi0 = na_hi0;
    a_lo1 = na_lo1; a_hi1 = na_hi1;
  }

#pragma unroll
  for (int mt = 0; mt < 2; ++mt) {
#pragma unroll
    for (int t = 0; t < 4; ++t) {
      const int n = n0 + t * 16 + mr;
      const float bv = bias[n];
#pragma unroll
      for (int r = 0; r < 8; ++r) {
        float v = acc[mt][t][r] + bv;
        if (do_relu) v = fmaxf(v, 0.f);
        Out[(size_t)(m0 + mt * 16 + r + half * 8) * N + n] = f2bf(v);
      }
    }
  }
}

// ---------------- final layer: out = y1 @ W2 + b2 + U (fp32 out, N=16) ----------------

__global__ __launch_bounds__(256) void layer2_wmma(
    const bf16* __restrict__ A, const bf16* __restrict__ W2p,
    const float* __restrict__ b2, const float* __restrict__ U,
    float* __restrict__ Out, int M, int K)
{
  const int lane = threadIdx.x & 31;
  const int wave = threadIdx.x >> 5;
  const int half = lane >> 4;
  const int mr   = lane & 15;
  const int m0   = (blockIdx.x * 8 + wave) * 16;

  v8f acc = {};
  const bf16* arow = A + (size_t)(m0 + mr) * K + half * 8;
  for (int kt = 0; kt < (K >> 5); ++kt) {
    const bf16* ap = arow + kt * 32;
    v8bf lo = *(const v8bf*)(ap);
    v8bf hi = *(const v8bf*)(ap + 16);
    v16bf afrag;
#pragma unroll
    for (int e = 0; e < 8; ++e) { afrag[e] = lo[e]; afrag[8 + e] = hi[e]; }
    v16bf bfrag = *(const v16bf*)(W2p + (((size_t)kt * 32 + lane) << 4));
    acc = wmma_bf16(afrag, bfrag, acc);
  }
  const float bb = b2[mr];
#pragma unroll
  for (int r = 0; r < 8; ++r) {
    int m = m0 + r + half * 8;
    Out[(size_t)m * NOUT + mr] = acc[r] + bb + U[(size_t)m * NOUT + mr];
  }
}

// ---------------- per-feature uni MLPs, accumulated in WMMA C registers ----------------
// U[b, :] = sum_i ( relu(relu(x[b,i]*uw1[i]+ub1[i]) @ uw2[i] + ub2[i]) @ uw3[i] ) + sum_i ub3[i]

__global__ __launch_bounds__(256) void uni_wmma(
    const float* __restrict__ x,   const float* __restrict__ uw1,
    const float* __restrict__ ub1, const float* __restrict__ ub2,
    const bf16* __restrict__ uw2p, const bf16* __restrict__ uw3p,
    const float* __restrict__ ub3s, float* __restrict__ U)
{
  __shared__ bf16 lds[8][16 * 32];   // per-wave 16x32 bf16 staging tile (D->A relayout)

  const int lane = threadIdx.x & 31;
  const int wave = threadIdx.x >> 5;
  const int half = lane >> 4;
  const int mr   = lane & 15;
  const int m0   = (blockIdx.x * 8 + wave) * 16;
  const int m    = m0 + mr;

  bf16* lt = &lds[wave][0];
  v8f uacc = {};

  for (int i = 0; i < NINP; ++i) {
    const float xv = x[(size_t)m * NINP + i];

    // h1 in A-fragment layout: elem e<8 -> k = half*8+e ; e>=8 -> k = half*8+16+(e-8)
    const f32x4* w1v = (const f32x4*)(uw1 + (size_t)i * UHU);
    const f32x4* b1v = (const f32x4*)(ub1 + (size_t)i * UHU);
    f32x4 wl0 = w1v[half * 2],     wl1 = w1v[half * 2 + 1];
    f32x4 wh0 = w1v[half * 2 + 4], wh1 = w1v[half * 2 + 5];
    f32x4 bl0 = b1v[half * 2],     bl1 = b1v[half * 2 + 1];
    f32x4 bh0 = b1v[half * 2 + 4], bh1 = b1v[half * 2 + 5];
    v16bf h1;
#pragma unroll
    for (int e = 0; e < 4; ++e) {
      h1[e]      = f2bf(fmaxf(fmaf(xv, wl0[e], bl0[e]), 0.f));
      h1[4 + e]  = f2bf(fmaxf(fmaf(xv, wl1[e], bl1[e]), 0.f));
      h1[8 + e]  = f2bf(fmaxf(fmaf(xv, wh0[e], bh0[e]), 0.f));
      h1[12 + e] = f2bf(fmaxf(fmaf(xv, wh1[e], bh1[e]), 0.f));
    }

    // h2pre = h1 @ uw2[i] : 16x32 x 32x32 -> two 16x16 f32 tiles
    const v16bf* wp = (const v16bf*)(uw2p + (((size_t)i * 2 * 32 + lane) << 4));
    v8f c0 = {}, c1 = {};
    c0 = wmma_bf16(h1, wp[0],  c0);
    c1 = wmma_bf16(h1, wp[32], c1);

    // bias + relu, stash as plain 16x32 bf16 tile in LDS (D layout: m = r + 8*half, n = mr)
    const float bc0 = ub2[(size_t)i * UHU + mr];
    const float bc1 = ub2[(size_t)i * UHU + 16 + mr];
#pragma unroll
    for (int r = 0; r < 8; ++r) {
      int mm = r + half * 8;
      lt[mm * 32 + mr]      = f2bf(fmaxf(c0[r] + bc0, 0.f));
      lt[mm * 32 + 16 + mr] = f2bf(fmaxf(c1[r] + bc1, 0.f));
    }
    // DS ops from one wave complete in order; barrier only stops compiler reordering.
    asm volatile("" ::: "memory");

    // reload in A-fragment layout
    const bf16* lrow = lt + mr * 32 + half * 8;
    v16bf h2;
#pragma unroll
    for (int e = 0; e < 8; ++e) { h2[e] = lrow[e]; h2[8 + e] = lrow[16 + e]; }
    asm volatile("" ::: "memory");

    // accumulate u += h2 @ uw3[i] across all features in C registers
    v16bf b3 = *(const v16bf*)(uw3p + (((size_t)i * 32 + lane) << 4));
    uacc = wmma_bf16(h2, b3, uacc);
  }

  const float ub = ub3s[mr];
#pragma unroll
  for (int r = 0; r < 8; ++r) {
    int mm = m0 + r + half * 8;
    U[(size_t)mm * NOUT + mr] = uacc[r] + ub;
  }
}

// ---------------- launch ----------------

extern "C" void kernel_launch(void* const* d_in, const int* in_sizes, int n_in,
                              void* d_out, int out_size, void* d_ws, size_t ws_size,
                              hipStream_t stream) {
  const float* x   = (const float*)d_in[0];
  const float* W0  = (const float*)d_in[1];
  const float* b0  = (const float*)d_in[2];
  const float* W1  = (const float*)d_in[3];
  const float* b1  = (const float*)d_in[4];
  const float* W2  = (const float*)d_in[5];
  const float* b2  = (const float*)d_in[6];
  const float* uw1 = (const float*)d_in[7];
  const float* ub1 = (const float*)d_in[8];
  const float* uw2 = (const float*)d_in[9];
  const float* ub2 = (const float*)d_in[10];
  const float* uw3 = (const float*)d_in[11];
  const float* ub3 = (const float*)d_in[12];
  float* out = (float*)d_out;

  // workspace carve-up (~76.3 MB total)
  char* p = (char*)d_ws;
  bf16*  xbf  = (bf16*)p;  p += (size_t)BATCH * NINP * 2;
  bf16*  y0   = (bf16*)p;  p += (size_t)BATCH * HID * 2;
  bf16*  y1   = (bf16*)p;  p += (size_t)BATCH * HID * 2;
  bf16*  W0p  = (bf16*)p;  p += (size_t)NINP * HID * 2;
  bf16*  W1p  = (bf16*)p;  p += (size_t)HID * HID * 2;
  bf16*  W2p  = (bf16*)p;  p += (size_t)HID * NOUT * 2;
  bf16*  uw2p = (bf16*)p;  p += (size_t)NINP * UHU * UHU * 2;
  bf16*  uw3p = (bf16*)p;  p += (size_t)NINP * UHU * NOUT * 2;
  float* Usum = (float*)p; p += (size_t)BATCH * NOUT * 4;
  float* ub3s = (float*)p; p += NOUT * 4;

  // 1) precision conversion + fragment packing
  {
    int n = BATCH * NINP;
    cvt_bf16_kernel<<<(n + 255) / 256, 256, 0, stream>>>(x, xbf, n);
  }
  {
    int t = NINP * HID;
    pack_b_frag_kernel<<<(t + 255) / 256, 256, 0, stream>>>(W0, W0p, NINP, HID, t);
  }
  {
    int t = HID * HID;
    pack_b_frag_kernel<<<(t + 255) / 256, 256, 0, stream>>>(W1, W1p, HID, HID, t);
  }
  {
    int t = HID * NOUT;
    pack_b_frag_kernel<<<(t + 255) / 256, 256, 0, stream>>>(W2, W2p, HID, NOUT, t);
  }
  {
    int t = NINP * UHU * UHU;   // batched over features
    pack_b_frag_kernel<<<(t + 255) / 256, 256, 0, stream>>>(uw2, uw2p, UHU, UHU, t);
  }
  {
    int t = NINP * UHU * NOUT;  // batched over features
    pack_b_frag_kernel<<<(t + 255) / 256, 256, 0, stream>>>(uw3, uw3p, UHU, NOUT, t);
  }
  ub3_sum_kernel<<<1, NOUT, 0, stream>>>(ub3, ub3s);

  // 2) main MLP hidden layers (bf16 WMMA, relu fused, async-LDS B staging)
  {
    dim3 grid(BATCH / 256, HID / 64);
    gemm_bf16_wmma<<<grid, 256, 0, stream>>>(xbf, W0p, b0, y0, BATCH, HID, NINP, 1);
    gemm_bf16_wmma<<<grid, 256, 0, stream>>>(y0, W1p, b1, y1, BATCH, HID, HID, 1);
  }

  // 3) per-feature uni MLPs -> Usum
  uni_wmma<<<BATCH / (16 * 8), 256, 0, stream>>>(x, uw1, ub1, ub2, uw2p, uw3p, ub3s, Usum);

  // 4) final projection + combine
  layer2_wmma<<<BATCH / (16 * 8), 256, 0, stream>>>(y1, W2p, b2, Usum, out, BATCH, HID);
}
